// RelPosEmb_34153579938286
// MI455X (gfx1250) — matrix operand order repaired and checked by reading, not compiled, not run on hardware
//
#include <hip/hip_runtime.h>
#include <hip/hip_bf16.h>

typedef __attribute__((ext_vector_type(16))) __bf16 v16bf;
typedef __attribute__((ext_vector_type(8)))  __bf16 v8bf;
typedef __attribute__((ext_vector_type(8)))  float  v8f;

#define NS      9
#define HDIM    64
#define WDIM    64
#define DDIM    64
#define RVALID  (WDIM + NS - 1)   // 72 relative positions actually used
#define RPAD    80                // padded to 5 N-tiles of 16
#define KTOT    128               // concat K: 64 (rel_w term) + 64 (rel_h term)
#define A_PITCH 136               // 128 bf16 + 8 pad -> 272B rows, conflict-free b128 reads
#define C_PITCH 84                // 80 f32 + 4 pad

// ---------------------------------------------------------------------------
// Setup: build concatenated B table in workspace, bf16, layout [r][k] with k
// contiguous (so a WMMA B-fragment lane reads two runs of 8 consecutive k per column r).
// k <  64 -> resized rel_width[r][k]      (term 1)
// k >= 64 -> resized rel_height[r][k-64]  (term 2)
// r in [72,80) -> zero padding (never gathered).
// ---------------------------------------------------------------------------
__global__ void relpos_build_B(const float* __restrict__ relH,
                               const float* __restrict__ relW,
                               __bf16* __restrict__ Bws) {
  int idx = blockIdx.x * blockDim.x + threadIdx.x;
  if (idx >= RPAD * KTOT) return;
  int r = idx >> 7;           // / KTOT
  int k = idx & (KTOT - 1);
  float val = 0.0f;
  const int in_size = RVALID;          // table rows == resize target here
  if (r < in_size) {
    const float* tab = (k < DDIM) ? relW : relH;
    int d = k & (DDIM - 1);
    // bilinear resize along axis 0 (identity when out==in; kept general)
    const float scale = (float)in_size / (float)in_size;
    float src = fmaxf(((float)r + 0.5f) * scale - 0.5f, 0.0f);
    float fl  = floorf(src);
    float w1  = src - fl;
    int i0 = (int)fl;
    if (i0 < 0) i0 = 0;
    if (i0 > in_size - 1) i0 = in_size - 1;
    int i1 = (i0 + 1 > in_size - 1) ? (in_size - 1) : (i0 + 1);
    val = tab[i0 * DDIM + d] * (1.0f - w1) + tab[i1 * DDIM + d] * w1;
  }
  Bws[idx] = (__bf16)val;
}

// ---------------------------------------------------------------------------
// Main: one block per (b*heads+hh, x). 128 threads = 4 waves.
// GEMM: M=64 (y), N=80 (r, padded), K=128, via v_wmma_f32_16x16x32_bf16.
// Then out[y][j] = C[y][63 - y + j].
// ---------------------------------------------------------------------------
__global__ __launch_bounds__(128)
void relpos_wmma_kernel(const float* __restrict__ q,
                        const __bf16* __restrict__ Bws,
                        float* __restrict__ out) {
  __shared__ __bf16 Alds[HDIM * A_PITCH];   // 17408 B
  __shared__ float  Clds[HDIM * C_PITCH];   // 21504 B

  const int bid = blockIdx.x;
  const int x   = bid & (HDIM - 1);
  const int bh  = bid >> 6;                 // b*heads + hh
  const int tid = threadIdx.x;

  const float* qbh = q + (size_t)bh * (HDIM * WDIM * DDIM);
  const float* A1  = qbh + (size_t)x * (WDIM * DDIM); // row y: stride DDIM (contig d)
  const float* A2  = qbh + (size_t)x * DDIM;          // row y: stride WDIM*DDIM

  // ---- stage A = [A1 | A2] into LDS as bf16 (vectorized f32x4 loads) ----
  for (int e = tid; e < HDIM * (DDIM / 4); e += 128) {
    int row = e >> 4, c4 = e & 15;
    float4 v = ((const float4*)(A1 + (size_t)row * DDIM))[c4];
    __bf16* dst = &Alds[row * A_PITCH + c4 * 4];
    dst[0] = (__bf16)v.x; dst[1] = (__bf16)v.y;
    dst[2] = (__bf16)v.z; dst[3] = (__bf16)v.w;
  }
  for (int e = tid; e < HDIM * (DDIM / 4); e += 128) {
    int row = e >> 4, c4 = e & 15;
    float4 v = ((const float4*)(A2 + (size_t)row * (WDIM * DDIM)))[c4];
    __bf16* dst = &Alds[row * A_PITCH + DDIM + c4 * 4];
    dst[0] = (__bf16)v.x; dst[1] = (__bf16)v.y;
    dst[2] = (__bf16)v.z; dst[3] = (__bf16)v.w;
  }
  __syncthreads();

  const int wave = tid >> 5;     // M-tile index 0..3
  const int lane = tid & 31;
  const int n16  = lane & 15;
  const int half = lane >> 4;

  v8f acc[5];
  #pragma unroll
  for (int nt = 0; nt < 5; ++nt)
    acc[nt] = (v8f){0.f, 0.f, 0.f, 0.f, 0.f, 0.f, 0.f, 0.f};

  // A fragment source row for this lane (row M = lane%16 within the M-tile)
  const __bf16* arow = &Alds[(wave * 16 + n16) * A_PITCH];

  #pragma unroll
  for (int s = 0; s < 4; ++s) {              // K step: K = 32*s .. 32*s+31
    // ISA 16-bit A 16x32 layout: half h -> VGPR0-3 = K[8h..8h+7], VGPR4-7 = K[16+8h..]
    const int kb = 32 * s + half * 8;
    v8bf alo = *(const v8bf*)(arow + kb);
    v8bf ahi = *(const v8bf*)(arow + kb + 16);
    v16bf afrag = __builtin_shufflevector(alo, ahi,
        0, 1, 2, 3, 4, 5, 6, 7, 8, 9, 10, 11, 12, 13, 14, 15);

    #pragma unroll
    for (int nt = 0; nt < 5; ++nt) {
      // ISA 16-bit B 32x16 layout: lane col = lane%16, K[8h..8h+7] then K[16+8h..16+8h+7]
      const __bf16* bp = Bws + (size_t)(nt * 16 + n16) * KTOT + 32 * s + half * 8;
      v8bf blo = *(const v8bf*)(bp);
      v8bf bhi = *(const v8bf*)(bp + 16);
      v16bf bfrag = __builtin_shufflevector(blo, bhi,
          0, 1, 2, 3, 4, 5, 6, 7, 8, 9, 10, 11, 12, 13, 14, 15);

      acc[nt] = __builtin_amdgcn_wmma_f32_16x16x32_bf16(
          /*neg_a=*/false, afrag, /*neg_b=*/false, bfrag,
          /*c_mod=*/(short)0, acc[nt], /*reuse_a=*/false, /*reuse_b=*/false);
    }
  }

  // ---- spill C tiles to LDS: lane holds col n16, rows = wave*16 + 8*half + i ----
  #pragma unroll
  for (int nt = 0; nt < 5; ++nt) {
    #pragma unroll
    for (int i = 0; i < 8; ++i) {
      Clds[(wave * 16 + half * 8 + i) * C_PITCH + nt * 16 + n16] = acc[nt][i];
    }
  }
  __syncthreads();

  // ---- diagonal-band gather: out[y][j] = C[y][63 - y + j] ----
  float* outp = out + (size_t)(bh * HDIM + x) * (WDIM * NS);
  for (int e = tid; e < WDIM * NS; e += 128) {
    int y = e / NS;
    int j = e - y * NS;
    outp[e] = Clds[y * C_PITCH + (WDIM - 1 - y + j)];
  }
}

// ---------------------------------------------------------------------------
extern "C" void kernel_launch(void* const* d_in, const int* in_sizes, int n_in,
                              void* d_out, int out_size, void* d_ws, size_t ws_size,
                              hipStream_t stream) {
  const float* q    = (const float*)d_in[0];
  // d_in[1] = H, d_in[2] = W (device scalars; shapes recovered from in_sizes)
  const float* relH = (const float*)d_in[3];
  const float* relW = (const float*)d_in[4];
  float*  out = (float*)d_out;
  __bf16* Bws = (__bf16*)d_ws;   // 80*128*2 = 20480 bytes

  const int BH = in_sizes[0] / (HDIM * WDIM * DDIM);   // B*heads = 128

  relpos_build_B<<<(RPAD * KTOT + 255) / 256, 256, 0, stream>>>(relH, relW, Bws);
  relpos_wmma_kernel<<<BH * HDIM, 128, 0, stream>>>(q, Bws, out);
}
